// TritonModel_12369505812696
// MI455X (gfx1250) — matrix operand order repaired and checked by reading, not compiled, not run on hardware
//
#include <hip/hip_runtime.h>

// DeltaNet fused recurrent decode step for MI455X (gfx1250, wave32).
// B=128, H=48, DK=DV=128. One workgroup (256 thr = 8 waves) per (b,h) pair.
// Memory-bound: 201MB state read (~8.6us at 23.3TB/s floor).
//
// s1 = bf16(state*decay) staged once in LDS in a K-pair-packed dword layout
// that IS the v_wmma_f32_16x16x32_bf16 B-operand register image.
// Both matvecs (k^T s1 and q^T s1) share the same B loads in one chunk loop.
// The rank-1 readout uses the identity  q^T(s1 + k (x) delta) = q^T s1 + (q.k)*delta,
// so the updated state is never constructed. (This skips the reference's
// bf16 rounding of the intermediate s2 -- a non-output storage artifact;
// deviation is ~2^-9/elem averaged over a 128-dot, far below the bf16
// output quantum, and strictly more accurate. s1 rounding, which feeds back
// through delta, is kept bit-faithful.)

#define NB 128
#define NH 48
#define DK 128
#define DV 128
#define PST 132   // padded LDS row stride in DWORDs (bank = 4*m + col mod 64)

typedef __attribute__((ext_vector_type(16))) __bf16       v16bf;
typedef __attribute__((ext_vector_type(8)))  float        v8f;
typedef __attribute__((ext_vector_type(8)))  unsigned int v8u;
typedef __attribute__((ext_vector_type(2)))  float        v2f;
typedef __attribute__((ext_vector_type(2)))  __bf16       v2bf;

__device__ __forceinline__ float bf2f(unsigned short b) {
  return __builtin_bit_cast(float, ((unsigned int)b) << 16);
}
// packed f32x2 -> bf16x2 (RNE)
__device__ __forceinline__ unsigned int pack2_bf16(float lo, float hi) {
  v2f f = {lo, hi};
  v2bf r = __builtin_convertvector(f, v2bf);
  return __builtin_bit_cast(unsigned int, r);
}
// fma(x,decay,-0) == x*decay bit-exactly (incl. signed zeros)
__device__ __forceinline__ float decay1(__bf16 x, float decay) {
  return fmaf((float)x, decay, -0.0f);
}

extern "C" __global__ __launch_bounds__(256)
void deltanet_decode_kernel(const unsigned short* __restrict__ q,
                            const unsigned short* __restrict__ k,
                            const unsigned short* __restrict__ v,
                            const unsigned short* __restrict__ beta,
                            const unsigned short* __restrict__ gate,
                            const unsigned short* __restrict__ state,
                            unsigned short* __restrict__ out) {
  // sP[m*PST + col] = ( s1[2m][col] , s1[2m+1][col] )  packed bf16 pair
  __shared__ unsigned int sP[(DK / 2) * PST];
  __shared__ unsigned int sQp[DK / 2];   // q as packed bf16 pairs
  __shared__ unsigned int sKp[DK / 2];   // k as packed bf16 pairs
  __shared__ unsigned int sVp[DV / 2];   // v as packed bf16 pairs
  __shared__ float        sKf[DK];       // k as f32
  __shared__ float        sQf[DK];       // q as f32

  const int bh = blockIdx.x;             // 0 .. B*H-1
  const int t  = threadIdx.x;            // 0 .. 255

  const unsigned short* Sg = state + (size_t)bh * (DK * DV);
  const float decay = expf(bf2f(gate[bh]));   // uniform scalars
  const float betaf = bf2f(beta[bh]);

  if (t < DK / 2) {
    unsigned int kw = ((const unsigned int*)k)[bh * (DK / 2) + t];
    unsigned int qw = ((const unsigned int*)q)[bh * (DK / 2) + t];
    v2bf kp = __builtin_bit_cast(v2bf, kw);
    v2bf qp = __builtin_bit_cast(v2bf, qw);
    sQp[t] = qw;
    sKp[t] = kw;
    sVp[t] = ((const unsigned int*)v)[bh * (DV / 2) + t];
    sKf[2 * t]     = (float)kp[0];
    sKf[2 * t + 1] = (float)kp[1];
    sQf[2 * t]     = (float)qp[0];
    sQf[2 * t + 1] = (float)qp[1];
  }

  // Stage s1 tile in pair-packed layout: each unit covers rows {2m,2m+1},
  // cols [c0,c0+8). Two coalesced 16B loads; v_cvt_pk_bf16_f32 directly
  // produces the cross-row packed dword (no interleave shuffles).
  #pragma unroll
  for (int i = 0; i < 4; ++i) {
    int u  = t + i * 256;                // 0 .. 1023
    int m  = u >> 4;                     // row pair 0..63
    int c0 = (u & 15) * 8;               // column group
    const unsigned short* p0 = Sg + m * 256 + c0;
    uint4 d0 = *(const uint4*)p0;            // row 2m   (8 bf16)
    uint4 d1 = *(const uint4*)(p0 + DV);     // row 2m+1 (8 bf16)
    v2bf a0 = __builtin_bit_cast(v2bf, d0.x), b0 = __builtin_bit_cast(v2bf, d1.x);
    v2bf a1 = __builtin_bit_cast(v2bf, d0.y), b1 = __builtin_bit_cast(v2bf, d1.y);
    v2bf a2 = __builtin_bit_cast(v2bf, d0.z), b2 = __builtin_bit_cast(v2bf, d1.z);
    v2bf a3 = __builtin_bit_cast(v2bf, d0.w), b3 = __builtin_bit_cast(v2bf, d1.w);
    unsigned int* dst = &sP[m * PST + c0];   // 16B aligned
    ((uint4*)dst)[0] = make_uint4(
        pack2_bf16(decay1(a0[0], decay), decay1(b0[0], decay)),
        pack2_bf16(decay1(a0[1], decay), decay1(b0[1], decay)),
        pack2_bf16(decay1(a1[0], decay), decay1(b1[0], decay)),
        pack2_bf16(decay1(a1[1], decay), decay1(b1[1], decay)));
    ((uint4*)dst)[1] = make_uint4(
        pack2_bf16(decay1(a2[0], decay), decay1(b2[0], decay)),
        pack2_bf16(decay1(a2[1], decay), decay1(b2[1], decay)),
        pack2_bf16(decay1(a3[0], decay), decay1(b3[0], decay)),
        pack2_bf16(decay1(a3[1], decay), decay1(b3[1], decay)));
  }
  __syncthreads();

  const int lane    = t & 31;
  const int wave    = t >> 5;            // wave w owns v-cols [16w, 16w+16)
  const int halfsel = lane >> 4;         // 0: lanes 0-15, 1: lanes 16-31
  const int col     = wave * 16 + (lane & 15);

  // ---- fused chunk loop: acc = k^T s1 and qs = q^T s1, shared B loads ----
  v8f acc = {0.f, 0.f, 0.f, 0.f, 0.f, 0.f, 0.f, 0.f};
  v8f qs  = {0.f, 0.f, 0.f, 0.f, 0.f, 0.f, 0.f, 0.f};
  #pragma unroll
  for (int c = 0; c < 4; ++c) {
    v8u ak, aq, bu;
    const int adw = 16 * c + 8 * halfsel;  // A-operand dword base
    const int mb  = 16 * c + 8 * halfsel;  // B-operand pair-row base
    #pragma unroll
    for (int j = 0; j < 8; ++j) {
      const int ai = adw + j;
      ak[j] = sKp[ai];                     // broadcast-row A (all M = k)
      aq[j] = sQp[ai];                     // broadcast-row A (all M = q)
      bu[j] = sP[(mb + j) * PST + col];    // shared s1 operand
    }
    v16bf bb = __builtin_bit_cast(v16bf, bu);
    acc = __builtin_amdgcn_wmma_f32_16x16x32_bf16(
        false, __builtin_bit_cast(v16bf, ak), false, bb, (short)0, acc,
        false, false);
    qs = __builtin_amdgcn_wmma_f32_16x16x32_bf16(
        false, __builtin_bit_cast(v16bf, aq), false, bb, (short)0, qs,
        false, false);
  }

  // qk = q . k  (per-wave tree reduction; fixed order -> deterministic)
  float4 qv = *(const float4*)&sQf[lane * 4];
  float4 kv = *(const float4*)&sKf[lane * 4];
  float qk = qv.x * kv.x + qv.y * kv.y + qv.z * kv.z + qv.w * kv.w;
  #pragma unroll
  for (int mlane = 16; mlane >= 1; mlane >>= 1)
    qk += __shfl_xor(qk, mlane, 32);

  // delta = beta * (v - k^T s1);  out = q^T s1 + (q.k) * delta
  const float vcol  = bf2f(((const unsigned short*)sVp)[col]);
  const float delta = betaf * (vcol - acc[0]);
  const float outv  = fmaf(qk, delta, qs[0]);

  if (lane < 16) {
    __bf16 r = (__bf16)outv;               // f32 -> bf16 RNE
    out[bh * DV + col] = __builtin_bit_cast(unsigned short, r);
  }
}

extern "C" void kernel_launch(void* const* d_in, const int* in_sizes, int n_in,
                              void* d_out, int out_size, void* d_ws, size_t ws_size,
                              hipStream_t stream) {
  const unsigned short* q     = (const unsigned short*)d_in[0];
  const unsigned short* k     = (const unsigned short*)d_in[1];
  const unsigned short* v     = (const unsigned short*)d_in[2];
  const unsigned short* beta  = (const unsigned short*)d_in[3];
  const unsigned short* gate  = (const unsigned short*)d_in[4];
  const unsigned short* state = (const unsigned short*)d_in[5];
  unsigned short* out = (unsigned short*)d_out;

  dim3 grid(NB * NH);   // 6144 workgroups, one per (b,h)
  dim3 block(256);      // 8 wave32's
  deltanet_decode_kernel<<<grid, block, 0, stream>>>(q, k, v, beta, gate, state, out);
}
